// Deepseekv3MoE_68633577390442
// MI455X (gfx1250) — matrix operand order repaired and checked by reading, not compiled, not run on hardware
//
#include <hip/hip_runtime.h>
#include <hip/hip_bf16.h>
#include <math.h>

#define T_TOK 1024
#define HID   768
#define NEXP  64
#define TOPK  8
#define NGRP  8
#define TOPG  4

typedef __attribute__((ext_vector_type(16))) __bf16 v16bf;
typedef __attribute__((ext_vector_type(8)))  float  v8f;
typedef __attribute__((ext_vector_type(2)))  int    v2i_t;

#define GLOBAL_AS __attribute__((address_space(1)))
#define LDS_AS    __attribute__((address_space(3)))

union FragBF { unsigned int u[8]; v16bf v; };

// Optional CDNA5 async global->LDS path (ASYNCcnt-tracked), guarded so the
// kernel still compiles if the toolchain doesn't declare these builtins.
#if defined(__has_builtin)
#if __has_builtin(__builtin_amdgcn_global_load_async_to_lds_b64) && \
    __has_builtin(__builtin_amdgcn_s_wait_asynccnt)
#define USE_ASYNC_LDS 1
#endif
#endif
#ifndef USE_ASYNC_LDS
#define USE_ASYNC_LDS 0
#endif

// Packed fp32->bf16 convert (RNE) via the CDNA hardware instruction.
// dst[15:0] = bf16(a), dst[31:16] = bf16(b) -- single VALU op.
__device__ __forceinline__ unsigned int pack2bf(float a, float b) {
  unsigned int r;
  asm("v_cvt_pk_bf16_f32 %0, %1, %2" : "=v"(r) : "v"(a), "v"(b));
  return r;
}
__device__ __forceinline__ unsigned short f2bf(float a) {
  return (unsigned short)(pack2bf(a, 0.f) & 0xFFFFu);
}
__device__ __forceinline__ float fast_sigmoid(float x) {
  return __builtin_amdgcn_rcpf(1.f + __expf(-x));
}

// ---------------------------------------------------------------- routing ---
__global__ void zero_counts_kernel(int* counts) {
  if (threadIdx.x < NEXP) counts[threadIdx.x] = 0;
}

__global__ void router_kernel(const float* __restrict__ x,
                              const float* __restrict__ gw,
                              const float* __restrict__ ebias,
                              int* __restrict__ counts, int* __restrict__ lists,
                              float* __restrict__ coefs,
                              unsigned short* __restrict__ xb) {
  __shared__ float xrow[HID];
  __shared__ float sl[NEXP];
  int t = blockIdx.x;
  int tid = threadIdx.x;  // 64 threads
  for (int i = tid; i < HID; i += 64) {
    float v = x[(size_t)t * HID + i];
    xrow[i] = v;
    xb[(size_t)t * HID + i] = f2bf(v);  // one-time fp32->bf16 of activations
  }
  __syncthreads();
  {
    const float* wrow = gw + (size_t)tid * HID;
    float acc = 0.f;
    for (int k = 0; k < HID; ++k) acc = fmaf(xrow[k], wrow[k], acc);
    sl[tid] = acc;
  }
  __syncthreads();
  if (tid == 0) {
    float sc[NEXP], swb[NEXP];
    for (int e = 0; e < NEXP; ++e) {
      float s = fast_sigmoid(sl[e]);
      sc[e] = s; swb[e] = s + ebias[e];
    }
    float gsc[NGRP];
    for (int g = 0; g < NGRP; ++g) {  // top-2 sum per group
      float m1 = -1e30f, m2 = -1e30f;
      for (int j = 0; j < NEXP / NGRP; ++j) {
        float v = swb[g * (NEXP / NGRP) + j];
        if (v > m1) { m2 = m1; m1 = v; } else if (v > m2) m2 = v;
      }
      gsc[g] = m1 + m2;
    }
    bool gsel[NGRP]; for (int g = 0; g < NGRP; ++g) gsel[g] = false;
    for (int k = 0; k < TOPG; ++k) {
      float best = -1e30f; int bi = 0;
      for (int g = 0; g < NGRP; ++g)
        if (!gsel[g] && gsc[g] > best) { best = gsc[g]; bi = g; }
      gsel[bi] = true;
    }
    bool esel[NEXP]; for (int e = 0; e < NEXP; ++e) esel[e] = false;
    int eidx[TOPK]; float ssum = 0.f;
    for (int k = 0; k < TOPK; ++k) {
      float best = -1e30f; int bi = 0;
      for (int e = 0; e < NEXP; ++e) {
        if (!gsel[e / (NEXP / NGRP)] || esel[e]) continue;
        if (swb[e] > best) { best = swb[e]; bi = e; }
      }
      esel[bi] = true; eidx[k] = bi; ssum += sc[bi];
    }
    float scale = 2.5f / (ssum + 1e-20f);
    for (int k = 0; k < TOPK; ++k) {
      int e = eidx[k];
      int pos = atomicAdd(&counts[e], 1);
      lists[e * T_TOK + pos] = t;
      coefs[e * T_TOK + pos] = sc[e] * scale;
    }
  }
}

__global__ void scan_kernel(const int* __restrict__ counts, int* __restrict__ segs) {
  if (threadIdx.x == 0) {
    int a = 0;
    for (int e = 0; e < NEXP; ++e) { segs[e] = a; a += counts[e]; }
    segs[NEXP] = a;
  }
}

// A-tile copy: 32 rows x 32 bf16 (64B/row) from bf16 source into LDS.
// 256 threads: thread -> (row = tid>>3, qword = tid&7), 8 bytes each.
__device__ __forceinline__ void load_a_tile(unsigned int (*As)[16],
                                            const unsigned short* __restrict__ src_row,
                                            int kk, int tid) {
  int m = tid >> 3, q = tid & 7;
  const unsigned short* src = src_row + kk + 4 * q;
#if USE_ASYNC_LDS
  __builtin_amdgcn_global_load_async_to_lds_b64(
      (GLOBAL_AS v2i_t*)src,
      (LDS_AS v2i_t*)&As[m][2 * q], 0, 0);
#else
  uint2 v = *(const uint2*)src;
  As[m][2 * q]     = v.x;
  As[m][2 * q + 1] = v.y;
#endif
}

__device__ __forceinline__ void a_tile_wait() {
#if USE_ASYNC_LDS
  __builtin_amdgcn_s_wait_asynccnt(0);
#endif
}

// --------------------------------------------------------------- up GEMM ----
// act[row, :] = coef(row) * silu(x_row @ Wg^T) * (x_row @ Wu^T)   (bf16 out)
// block = 256 thr (8 wave32), tile = 32 rows x 128 cols, K step 32 (bf16 WMMA)
template <int IDIM, bool SHARED>
__global__ __launch_bounds__(256)
void up_kernel(const float* __restrict__ wg_base, const float* __restrict__ wu_base,
               const unsigned short* __restrict__ xb,
               const int* __restrict__ lists, const float* __restrict__ coefs,
               const int* __restrict__ counts, const int* __restrict__ segs,
               unsigned short* __restrict__ act) {
  int e = blockIdx.x, mtile = blockIdx.y, nchunk = blockIdx.z;
  int cnt = SHARED ? T_TOK : counts[e];
  if (mtile * 32 >= cnt) return;
  const float* wg = wg_base + (SHARED ? 0 : (size_t)e * IDIM * HID);
  const float* wu = wu_base + (SHARED ? 0 : (size_t)e * IDIM * HID);
  int segbase = SHARED ? 0 : segs[e];

  __shared__ unsigned int As[32][16];   // 32 rows x 32 K (bf16 pairs)
  __shared__ unsigned int Bg[128][16];  // 128 cols x 32 K
  __shared__ unsigned int Bu[128][16];
  __shared__ int   rowTok[32];
  __shared__ float rowCoef[32];

  int tid = threadIdx.x, lane = tid & 31, wv = tid >> 5;
  if (tid < 32) {
    int r = mtile * 32 + tid;
    if (SHARED) { rowTok[tid] = r; rowCoef[tid] = 1.f; }
    else {
      bool valid = r < cnt;
      // invalid tail rows read token 0: their WMMA output rows are never stored
      rowTok[tid]  = valid ? lists[e * T_TOK + r] : 0;
      rowCoef[tid] = valid ? coefs[e * T_TOK + r] : 0.f;
    }
  }
  __syncthreads();

  v8f accG0 = {}, accG1 = {}, accU0 = {}, accU1 = {};
  const int nbase = nchunk * 128;
  const unsigned short* arow = xb + (size_t)rowTok[tid >> 3] * HID;

  for (int kk = 0; kk < HID; kk += 32) {
    load_a_tile(As, arow, kk, tid);
    // B tiles: fp32 -> bf16, 16B global loads, 8 float4 per 128-col row
    for (int idx = tid; idx < 128 * 8; idx += 256) {
      int r = idx >> 3, q = idx & 7;
      const float* pg = wg + (size_t)(nbase + r) * HID + kk + 4 * q;
      const float* pu = wu + (size_t)(nbase + r) * HID + kk + 4 * q;
      float4 g4 = *(const float4*)pg;
      float4 u4 = *(const float4*)pu;
      if (kk + 32 < HID) __builtin_prefetch(pg + 32, 0, 0);  // next K chunk
      Bg[r][2 * q]     = pack2bf(g4.x, g4.y);
      Bg[r][2 * q + 1] = pack2bf(g4.z, g4.w);
      Bu[r][2 * q]     = pack2bf(u4.x, u4.y);
      Bu[r][2 * q + 1] = pack2bf(u4.z, u4.w);
    }
    a_tile_wait();
    __syncthreads();

    int h = lane >> 4, ml = lane & 15;
    FragBF a0, a1, bg, bu;
#pragma unroll
    for (int j = 0; j < 8; ++j) {
      int c = 8 * h + j;  // contiguous k = 16h + s, matches B
      a0.u[j] = As[ml][c];
      a1.u[j] = As[16 + ml][c];
      bg.u[j] = Bg[wv * 16 + ml][8 * h + j];   // contiguous k = 16h + s, same as A
      bu.u[j] = Bu[wv * 16 + ml][8 * h + j];
    }
    accG0 = __builtin_amdgcn_wmma_f32_16x16x32_bf16(false, a0.v, false, bg.v, (short)0, accG0, false, false);
    accG1 = __builtin_amdgcn_wmma_f32_16x16x32_bf16(false, a1.v, false, bg.v, (short)0, accG1, false, false);
    accU0 = __builtin_amdgcn_wmma_f32_16x16x32_bf16(false, a0.v, false, bu.v, (short)0, accU0, false, false);
    accU1 = __builtin_amdgcn_wmma_f32_16x16x32_bf16(false, a1.v, false, bu.v, (short)0, accU1, false, false);
    __syncthreads();
  }

  int h = lane >> 4, nl = lane & 15;
  int nglob = nbase + wv * 16 + nl;
#pragma unroll
  for (int v = 0; v < 8; ++v) {
    int m0 = v + 8 * h;                         // ISA f32 C/D layout
    int r0 = mtile * 32 + m0;
    if (r0 < cnt) {
      float g = accG0[v], u = accU0[v];
      float val = g * fast_sigmoid(g) * u * rowCoef[m0];
      act[(size_t)(segbase + r0) * IDIM + nglob] = f2bf(val);
    }
    int r1 = r0 + 16;
    if (r1 < cnt) {
      float g = accG1[v], u = accU1[v];
      float val = g * fast_sigmoid(g) * u * rowCoef[16 + m0];
      act[(size_t)(segbase + r1) * IDIM + nglob] = f2bf(val);
    }
  }
}

// ------------------------------------------------------------- down GEMM ----
// out[token, :] (+)= act_row @ W2^T ; SHARED: plain store (initializes out)
template <int KDIM, bool SHARED>
__global__ __launch_bounds__(256)
void down_kernel(const float* __restrict__ w2_base,
                 const unsigned short* __restrict__ act,
                 const int* __restrict__ lists, const int* __restrict__ counts,
                 const int* __restrict__ segs, float* __restrict__ out) {
  int e = blockIdx.x, mtile = blockIdx.y, nchunk = blockIdx.z;
  int cnt = SHARED ? T_TOK : counts[e];
  if (mtile * 32 >= cnt) return;
  const float* w2 = w2_base + (SHARED ? 0 : (size_t)e * HID * KDIM);
  int segbase = SHARED ? 0 : segs[e];

  __shared__ unsigned int As[32][16];
  __shared__ unsigned int Bs[128][16];
  __shared__ int rowTok[32];

  int tid = threadIdx.x, lane = tid & 31, wv = tid >> 5;
  if (tid < 32) {
    int r = mtile * 32 + tid;
    rowTok[tid] = SHARED ? r : ((r < cnt) ? lists[e * T_TOK + r] : 0);
  }
  __syncthreads();

  v8f acc0 = {}, acc1 = {};
  const int nbase = nchunk * 128;
  // act rows are contiguous (compacted); tail reads land in the padded region
  const unsigned short* arow =
      act + (size_t)(segbase + mtile * 32 + (tid >> 3)) * KDIM;

  for (int kk = 0; kk < KDIM; kk += 32) {
    load_a_tile(As, arow, kk, tid);
    for (int idx = tid; idx < 128 * 8; idx += 256) {
      int r = idx >> 3, q = idx & 7;
      const float* p = w2 + (size_t)(nbase + r) * KDIM + kk + 4 * q;
      float4 x4 = *(const float4*)p;
      if (kk + 32 < KDIM) __builtin_prefetch(p + 32, 0, 0);
      Bs[r][2 * q]     = pack2bf(x4.x, x4.y);
      Bs[r][2 * q + 1] = pack2bf(x4.z, x4.w);
    }
    a_tile_wait();
    __syncthreads();

    int h = lane >> 4, ml = lane & 15;
    FragBF a0, a1, b;
#pragma unroll
    for (int j = 0; j < 8; ++j) {
      int c = 8 * h + j;
      a0.u[j] = As[ml][c];
      a1.u[j] = As[16 + ml][c];
      b.u[j]  = Bs[wv * 16 + ml][8 * h + j];
    }
    acc0 = __builtin_amdgcn_wmma_f32_16x16x32_bf16(false, a0.v, false, b.v, (short)0, acc0, false, false);
    acc1 = __builtin_amdgcn_wmma_f32_16x16x32_bf16(false, a1.v, false, b.v, (short)0, acc1, false, false);
    __syncthreads();
  }

  int h = lane >> 4, nl = lane & 15;
  int nglob = nbase + wv * 16 + nl;
#pragma unroll
  for (int v = 0; v < 8; ++v) {
    int m0 = v + 8 * h;
    int r0 = mtile * 32 + m0, r1 = r0 + 16;
    if (r0 < cnt) {
      int t = rowTok[m0];
      if (SHARED) out[(size_t)t * HID + nglob] = acc0[v];
      else        atomicAdd(out + (size_t)t * HID + nglob, acc0[v]);
    }
    if (r1 < cnt) {
      int t = rowTok[16 + m0];
      if (SHARED) out[(size_t)t * HID + nglob] = acc1[v];
      else        atomicAdd(out + (size_t)t * HID + nglob, acc1[v]);
    }
  }
}

// ----------------------------------------------------------------- launch ---
extern "C" void kernel_launch(void* const* d_in, const int* in_sizes, int n_in,
                              void* d_out, int out_size, void* d_ws, size_t ws_size,
                              hipStream_t stream) {
  const float* x       = (const float*)d_in[0];
  const float* gate_w  = (const float*)d_in[1];
  const float* e_bias  = (const float*)d_in[2];
  const float* w1      = (const float*)d_in[3];
  const float* w3      = (const float*)d_in[4];
  const float* w2      = (const float*)d_in[5];
  const float* ws_gate = (const float*)d_in[6];
  const float* ws_up   = (const float*)d_in[7];
  const float* ws_down = (const float*)d_in[8];
  float* out = (float*)d_out;

  char* base = (char*)d_ws;
  int*   counts = (int*)(base);                                  // 64 ints
  int*   segs   = (int*)(base + 256);                            // 65 ints
  int*   lists  = (int*)(base + 1024);                           // 64*1024 ints
  float* coefs  = (float*)(base + 1024 + 64 * 1024 * 4);         // 64*1024 f32
  unsigned short* xb   = (unsigned short*)(base + 1024 + 2 * 64 * 1024 * 4); // 1024*768 bf16
  unsigned short* actR = xb + (size_t)1024 * 768;                // (8192+64)*384 bf16 (padded)
  unsigned short* actS = actR + (size_t)(8192 + 64) * 384;       // 1024*768 bf16

  zero_counts_kernel<<<1, 64, 0, stream>>>(counts);
  router_kernel<<<1024, 64, 0, stream>>>(x, gate_w, e_bias, counts, lists, coefs, xb);
  scan_kernel<<<1, 64, 0, stream>>>(counts, segs);

  // shared expert first: plain-store down initializes d_out
  up_kernel<768, true><<<dim3(1, 32, 6), 256, 0, stream>>>(
      ws_gate, ws_up, xb, lists, coefs, counts, segs, actS);
  down_kernel<768, true><<<dim3(1, 32, 6), 256, 0, stream>>>(
      ws_down, actS, lists, counts, segs, out);

  // routed experts: gathered top-8, coef folded into act, atomic scatter-add
  up_kernel<384, false><<<dim3(64, 32, 3), 256, 0, stream>>>(
      w1, w3, xb, lists, coefs, counts, segs, actR);
  down_kernel<384, false><<<dim3(64, 32, 6), 256, 0, stream>>>(
      w2, actR, lists, counts, segs, out);
}
